// RNN_85676007620770
// MI455X (gfx1250) — matrix operand order repaired and checked by reading, not compiled, not run on hardware
//
#include <hip/hip_runtime.h>

#define H      2048
#define TUNED  10
#define BB     16
#define TT     1024
#define NIN    8
#define NOUT   8
#define TAU    0.01f
#define TAU_FACIL 1.5f
#define P_REL  1.0f
#define BETA   1.5f
#define DT     0.001f

typedef __bf16 bf16_t;
typedef __attribute__((ext_vector_type(16))) __bf16       v16bf;
typedef __attribute__((ext_vector_type(8)))  float        v8f;
typedef __attribute__((ext_vector_type(4)))  unsigned int u32x4;

union V32B { u32x4 q[2]; v16bf bf; };   // trivial members -> default-constructible

__device__ __forceinline__ bf16_t f2bf(float f) {
    unsigned u = __builtin_bit_cast(unsigned, f);
    u += 0x7FFFu + ((u >> 16) & 1u);              // round to nearest even
    unsigned short s = (unsigned short)(u >> 16);
    return __builtin_bit_cast(bf16_t, s);
}

// ---- Precompute Wm = mask * concat(W_tuned, W_fixed) in bf16 (time-invariant) ----
__global__ void build_wm_kernel(const float* __restrict__ wt, const float* __restrict__ wf,
                                const float* __restrict__ mask, bf16_t* __restrict__ wm) {
    int idx = blockIdx.x * blockDim.x + threadIdx.x;
    if (idx >= H * H) return;
    int n = idx / H, k = idx % H;
    float w = (k < TUNED) ? wt[n * TUNED + k] : wf[n * (H - TUNED) + (k - TUNED)];
    wm[idx] = f2bf(mask[idx] * w);
}

// ---- Initialize recurrent state ----
__global__ void init_state_kernel(const float* __restrict__ h0,
                                  float* __restrict__ h_s, float* __restrict__ r_s,
                                  float* __restrict__ hT_s, bf16_t* __restrict__ act) {
    int idx = blockIdx.x * blockDim.x + threadIdx.x;
    if (idx >= BB * H) return;
    float h  = h0[idx];
    float th = tanhf(h);
    h_s[idx]  = h;
    r_s[idx]  = P_REL;
    hT_s[idx] = th;
    act[idx]  = f2bf(P_REL * th);   // r0 * tanh(h0)
}

// ---- z = tanh(h) @ W_hz.T  : [16,8] reduction over H (wave reduce) ----
__global__ void z_kernel(const float* __restrict__ hT_s, const float* __restrict__ W_hz,
                         float* __restrict__ z_state, float* __restrict__ out_z_t) {
    int b    = blockIdx.x;           // grid = B
    int o    = threadIdx.x >> 5;     // block = NOUT*32
    int lane = threadIdx.x & 31;
    const float* hp = hT_s + b * H;
    const float* wp = W_hz + o * H;
    float s = 0.f;
    #pragma unroll 8
    for (int j = lane; j < H; j += 32) s += hp[j] * wp[j];
    #pragma unroll
    for (int m = 16; m >= 1; m >>= 1) s += __shfl_xor(s, m, 32);
    if (lane == 0) {
        z_state[b * NOUT + o] = s;
        if (out_z_t) out_z_t[b * TT * NOUT + o] = s;
    }
}

// ---- One recurrent step: split-K x4 WMMA GEMM + fused state update ----
// grid = H/16 blocks (one 16x16 N-tile each), block = 128 threads = 4 waves
__global__ __launch_bounds__(128) void step_kernel(
    const bf16_t* __restrict__ wm, const bf16_t* __restrict__ act_in,
    bf16_t* __restrict__ act_out,
    float* __restrict__ h_s, float* __restrict__ r_s, float* __restrict__ hT_s,
    const float* __restrict__ x_t,      // base + t*NIN;  [b] stride = T*NIN
    const float* __restrict__ W_ih, const float* __restrict__ W_zh,
    const float* __restrict__ z_state, const float* __restrict__ tau_depr,
    float* __restrict__ out_h_t, float* __restrict__ out_r_t) {

    __shared__ float lds_acc[4][256];

    const int tid  = threadIdx.x;
    const int wave = tid >> 5;
    const int lane = tid & 31;
    const int n0   = blockIdx.x * 16;
    const int lh   = lane & 15;
    const int hi   = lane >> 4;
    const int kw   = wave * (H / 4);       // split-K: 512 per wave

    v8f acc = {0.f, 0.f, 0.f, 0.f, 0.f, 0.f, 0.f, 0.f};

    // A tile base: act row m = lh (batch), bf16 row-major [B][H]
    //   lanes 0-15 hold K {+0..7, +16..23}; lanes 16-31 hold K {+8..15, +24..31}
    const bf16_t* ap = act_in + lh * H + kw + hi * 8;
    // B tile base: need Wm^T -> lane's column n = row (n0+lh) of Wm, contiguous K
    //   lanes 0-15 hold K {+0..7, +16..23}; lanes 16-31 hold K {+8..15, +24..31}
    const bf16_t* bp = wm + (size_t)(n0 + lh) * H + kw + hi * 8;

    // Constant trip count, fully unrolled: all 64 loads fold into immediate
    // offsets off two base pointers; no per-iteration address math or exec-mask
    // loop overhead on the recurrent critical path.
    #pragma unroll
    for (int it = 0; it < 16; ++it) {
        V32B a, b;
        a.q[0] = *(const u32x4*)(ap + it * 32);
        a.q[1] = *(const u32x4*)(ap + it * 32 + 16);
        b.q[0] = *(const u32x4*)(bp + it * 32);
        b.q[1] = *(const u32x4*)(bp + it * 32 + 16);
        acc = __builtin_amdgcn_wmma_f32_16x16x32_bf16(
            /*neg_a=*/false, a.bf, /*neg_b=*/false, b.bf,
            /*c_mod=*/(short)0, acc, /*reuse_a=*/false, /*reuse_b=*/false);
    }

    #pragma unroll
    for (int i = 0; i < 8; ++i) lds_acc[wave][lane * 8 + i] = acc[i];
    __syncthreads();

    // Epilogue: 256 (b,j) outputs, 2 per thread; fused tanh + depression update
    #pragma unroll
    for (int s = tid; s < 256; s += 128) {
        float a = lds_acc[0][s] + lds_acc[1][s] + lds_acc[2][s] + lds_acc[3][s];
        const int lv = s >> 3, i = s & 7;
        const int m = ((lv >> 4) << 3) + i;     // batch index 0..15
        const int n = n0 + (lv & 15);           // hidden index

        float inp = 0.f, zw = 0.f;
        const float* xr = x_t + (size_t)m * TT * NIN;
        const float* wr = W_ih + n * NIN;
        #pragma unroll
        for (int k = 0; k < NIN; ++k) inp += xr[k] * wr[k];
        const float* zr = z_state + m * NOUT;
        const float* wz = W_zh + n * NOUT;
        #pragma unroll
        for (int o = 0; o < NOUT; ++o) zw += zr[o] * wz[o];

        const int idx = m * H + n;
        float h  = h_s[idx];
        float r  = r_s[idx];
        float th = hT_s[idx];

        float dh  = (-h + a + inp + zw) * (1.0f / TAU);
        float hn  = h + dh * DT;
        float thn = tanhf(hn);
        float rn  = r + ((P_REL - r) / tau_depr[n] - BETA * r * th) * DT;

        h_s[idx]  = hn;
        r_s[idx]  = rn;
        hT_s[idx] = thn;
        act_out[idx] = f2bf(rn * thn);

        out_h_t[(size_t)m * TT * H + n] = hn;
        out_r_t[(size_t)m * TT * H + n] = rn;
    }
}

extern "C" void kernel_launch(void* const* d_in, const int* in_sizes, int n_in,
                              void* d_out, int out_size, void* d_ws, size_t ws_size,
                              hipStream_t stream) {
    (void)in_sizes; (void)n_in; (void)out_size; (void)ws_size;

    const float* x    = (const float*)d_in[0];
    const float* h0   = (const float*)d_in[1];
    const float* W_ih = (const float*)d_in[2];
    const float* W_t  = (const float*)d_in[3];
    const float* W_f  = (const float*)d_in[4];
    const float* mask = (const float*)d_in[5];
    const float* W_hz = (const float*)d_in[6];
    const float* W_zh = (const float*)d_in[7];
    const float* tau  = (const float*)d_in[8];

    char* ws = (char*)d_ws;
    size_t off = 0;
    auto carve = [&](size_t bytes) -> void* {
        void* p = ws + off;
        off = (off + bytes + 255) & ~(size_t)255;
        return p;
    };
    bf16_t* wm      = (bf16_t*)carve((size_t)H * H * sizeof(bf16_t));   // 8 MB, L2-resident
    bf16_t* act0    = (bf16_t*)carve((size_t)BB * H * sizeof(bf16_t));  // ping
    bf16_t* act1    = (bf16_t*)carve((size_t)BB * H * sizeof(bf16_t));  // pong
    float*  h_s     = (float*)carve((size_t)BB * H * sizeof(float));
    float*  r_s     = (float*)carve((size_t)BB * H * sizeof(float));
    float*  hT_s    = (float*)carve((size_t)BB * H * sizeof(float));
    float*  z_state = (float*)carve((size_t)BB * NOUT * sizeof(float));

    float* out_z = (float*)d_out;                     // [B,T,NOUT]
    float* out_h = out_z + (size_t)BB * TT * NOUT;    // [B,T,H]
    float* out_r = out_h + (size_t)BB * TT * H;       // [B,T,H]

    build_wm_kernel<<<(H * H + 255) / 256, 256, 0, stream>>>(W_t, W_f, mask, wm);
    init_state_kernel<<<(BB * H + 255) / 256, 256, 0, stream>>>(h0, h_s, r_s, hT_s, act0);
    z_kernel<<<BB, NOUT * 32, 0, stream>>>(hT_s, W_hz, z_state, nullptr);  // z_{-1} from tanh(h0)

    for (int t = 0; t < TT; ++t) {
        const bf16_t* ain  = (t & 1) ? act1 : act0;
        bf16_t*       aout = (t & 1) ? act0 : act1;
        step_kernel<<<H / 16, 128, 0, stream>>>(
            wm, ain, aout, h_s, r_s, hT_s,
            x + (size_t)t * NIN, W_ih, W_zh, z_state, tau,
            out_h + (size_t)t * H, out_r + (size_t)t * H);
        z_kernel<<<BB, NOUT * 32, 0, stream>>>(hT_s, W_hz, z_state, out_z + (size_t)t * NOUT);
    }
}
